// GatedDeltaNet_69844758167941
// MI455X (gfx1250) — matrix operand order repaired and checked by reading, not compiled, not run on hardware
//
#include <hip/hip_runtime.h>
#include <cstdint>

typedef unsigned short u16;
typedef __attribute__((ext_vector_type(16))) __bf16 v16bf;
typedef __attribute__((ext_vector_type(8)))  float  v8f;
typedef __attribute__((ext_vector_type(4)))  unsigned int u32x4;
typedef __attribute__((ext_vector_type(8)))  int i32x8;
typedef __attribute__((ext_vector_type(4)))  int i32x4;
typedef __attribute__((ext_vector_type(4)))  int v4i;

#define DEV static __device__ __forceinline__
#define AS1 __attribute__((address_space(1)))
#define AS3 __attribute__((address_space(3)))

#ifndef __has_builtin
#define __has_builtin(x) 0
#endif
#if __has_builtin(__builtin_amdgcn_global_load_async_to_lds_b128)
#define HAVE_ASYNC 1
#else
#define HAVE_ASYNC 0
#endif
#if __has_builtin(__builtin_amdgcn_tensor_load_to_lds)
#define HAVE_TDM 1
#else
#define HAVE_TDM 0
#endif

DEV float bf2f(u16 h) { unsigned u = ((unsigned)h) << 16; float f; __builtin_memcpy(&f, &u, 4); return f; }
DEV u16 f2bf(float f) {
  unsigned u; __builtin_memcpy(&u, &f, 4);
  u += 0x7FFFu + ((u >> 16) & 1u);            // round-to-nearest-even
  return (u16)(u >> 16);
}

DEV void wait_async0() {
#if __has_builtin(__builtin_amdgcn_s_wait_asynccnt)
  __builtin_amdgcn_s_wait_asynccnt(0);
#else
  asm volatile("s_wait_asynccnt 0" ::: "memory");
#endif
}
DEV void wait_tensor0() {
#if __has_builtin(__builtin_amdgcn_s_wait_tensorcnt)
  __builtin_amdgcn_s_wait_tensorcnt(0);
#else
  asm volatile("s_wait_tensorcnt 0" ::: "memory");
#endif
}

#if HAVE_ASYNC
DEV void async_cp_b128(const void* g, void* l) {
  // signature (from hipcc diagnostic): (v4i addrspace(1)* src, v4i addrspace(3)* dst, imm off, imm cpol)
  __builtin_amdgcn_global_load_async_to_lds_b128((AS1 v4i*)g, (AS3 v4i*)l, 0, 0);
}
#endif

#if HAVE_TDM
DEV unsigned lds_off(const void* p) { return (unsigned)(size_t)(AS3 const void*)p; }

// 2-D TDM tile load: 16 rows x 128 bf16, global row stride 3072 elems,
// LDS rows padded via TDM pad (64 DW data + 4 DW pad = 136 u16 = qlds stride).
DEV void tdm_load_tile(unsigned lds_addr, const u16* gsrc) {
  unsigned long long ga = (unsigned long long)(size_t)gsrc;
  u32x4 g0;
  g0[0] = 1u;                                                // count=1, user desc
  g0[1] = lds_addr;                                          // lds_addr [63:32]
  g0[2] = (unsigned)ga;                                      // global_addr lo
  g0[3] = (unsigned)((ga >> 32) & 0x01FFFFFFu) | (2u << 30); // addr hi | type=2
  i32x8 g1;
  // data_size=1(2B) @16, pad_enable @20, pad_interval=5(64DW) @22, pad_amount=3(4DW) @25
  g1[0] = (int)((1u << 16) | (1u << 20) | (5u << 22) | (3u << 25));
  g1[1] = (int)(3072u << 16);            // tensor_dim0 = 3072 (bits 79:48, low part)
  g1[2] = 0;                             // dim0 hi, tensor_dim1 lo16 (dim1 = 1<<20)
  g1[3] = (int)(0x0010u | (128u << 16)); // tensor_dim1 hi16, tile_dim0 = 128
  g1[4] = 16;                            // tile_dim1 = 16, tile_dim2 = 0
  g1[5] = 3072;                          // tensor_dim0_stride (48b, low part)
  g1[6] = 0; g1[7] = 0;
  i32x4 g2 = {0, 0, 0, 0};
  i32x4 g3 = {0, 0, 0, 0};
#if __has_include(<hip/amd_detail/amd_gfx1250_TDM.h>)
  i32x8 g4 = {0, 0, 0, 0, 0, 0, 0, 0};
  __builtin_amdgcn_tensor_load_to_lds(g0, g1, g2, g3, g4, 0);   // 6-arg toolchain
#else
  __builtin_amdgcn_tensor_load_to_lds(g0, g1, g2, g3, 0);       // 5-arg toolchain
#endif
}
#endif

union FragU { v16bf v; unsigned u[8]; };

// A-fragment (16x32 bf16, ISA 7.12.2): VGPR v: K = (v>=4?16:0) + (lane/16)*8 + (v&3)*2
DEV v16bf ld_fragA(const u16* base, int stride, int row0) {
  int lane = threadIdx.x & 31;
  const u16* p = base + (size_t)(row0 + (lane & 15)) * stride + ((lane >> 4) << 3);
  FragU f;
#pragma unroll
  for (int v = 0; v < 8; ++v) {
    int k = ((v & 4) << 2) + ((v & 3) << 1);
    f.u[v] = *(const unsigned*)(p + k);
  }
  return f.v;
}

// B-fragment (32x16 bf16): VGPR v: K = (v>=4?16:0) + (lane/16)*8 + (v&3)*2 (same as A).
// LDS tile stored as [N][K] row-major.
DEV v16bf ld_fragB(const u16* base, int stride, int row0) {
  int lane = threadIdx.x & 31;
  const u16* p = base + (size_t)(row0 + (lane & 15)) * stride + ((lane >> 4) << 3);
  FragU f;
#pragma unroll
  for (int v = 0; v < 8; ++v) f.u[v] = *(const unsigned*)(p + ((v & 4) << 2) + ((v & 3) << 1));
  return f.v;
}

DEV v8f wmma_bf16(v16bf a, v16bf b, v8f c) {
  return __builtin_amdgcn_wmma_f32_16x16x32_bf16(false, a, false, b, (short)0, c, false, false);
}

// ---------------------------------------------------------------- convert f32 -> bf16
__global__ void f2bf_kernel(const float* __restrict__ in, u16* __restrict__ out, size_t n) {
  size_t i = (size_t)blockIdx.x * blockDim.x + threadIdx.x;
  if (i < n) out[i] = f2bf(in[i]);
}

// ---------------------------------------------------------------- bf16 WMMA GEMM
// C[M,N] = A[M,K] @ B[K,N]. Block tile 128x128, BK=32; 8 waves = 2(m) x 4(n).
template <bool BF16OUT>
__global__ __launch_bounds__(256) void gemm_wmma_kernel(
    const u16* __restrict__ A, const u16* __restrict__ B, void* __restrict__ Cout,
    int M, int N, int K)
{
  __shared__ __align__(16) u16 As[128][40];
  __shared__ __align__(16) u16 Bs[128][40];   // Bs[n][k] = B[k][n]
  const int t = threadIdx.x;
  const int wave = t >> 5, lane = t & 31;
  const int wm = wave >> 2, wn = wave & 3;
  const int m0 = blockIdx.y * 128, n0 = blockIdx.x * 128;

  v8f acc[4][2] = {};

  for (int kb = 0; kb < K; kb += 32) {
#if HAVE_ASYNC
    {                                                  // A tile via async global->LDS B128
      int r = t >> 1;
      int cbase = (t & 1) * 16;
#pragma unroll
      for (int p = 0; p < 2; ++p) {
        int c = cbase + p * 8;
        async_cp_b128(A + (size_t)(m0 + r) * K + kb + c, &As[r][c]);
      }
    }
#else
#pragma unroll
    for (int p = 0; p < 4; ++p) {                      // A tile: 128x32
      int r = p * 32 + (t >> 3);
      int c = (t & 7) * 4;
      *(uint2*)&As[r][c] = *(const uint2*)(A + (size_t)(m0 + r) * K + kb + c);
    }
#endif
    {                                                  // B tile (transposed into LDS)
      int k = t & 31, ng = t >> 5;
      const u16* g = B + (size_t)(kb + k) * N + n0 + ng * 16;
#pragma unroll
      for (int i = 0; i < 16; ++i) Bs[ng * 16 + i][k] = g[i];
    }
    if (kb + 32 < K)
      __builtin_prefetch(A + (size_t)(m0 + (t >> 3)) * K + kb + 32, 0, 1);
#if HAVE_ASYNC
    wait_async0();
#endif
    __syncthreads();

    v16bf af[4], bfr[2];
#pragma unroll
    for (int i = 0; i < 4; ++i) af[i] = ld_fragA(&As[0][0], 40, wm * 64 + i * 16);
#pragma unroll
    for (int j = 0; j < 2; ++j) bfr[j] = ld_fragB(&Bs[0][0], 40, wn * 32 + j * 16);
#pragma unroll
    for (int i = 0; i < 4; ++i)
#pragma unroll
      for (int j = 0; j < 2; ++j)
        acc[i][j] = wmma_bf16(af[i], bfr[j], acc[i][j]);
    __syncthreads();
  }
  const int half = lane >> 4, nn = lane & 15;
#pragma unroll
  for (int i = 0; i < 4; ++i)
#pragma unroll
    for (int j = 0; j < 2; ++j)
#pragma unroll
      for (int r = 0; r < 8; ++r) {
        int row = m0 + wm * 64 + i * 16 + r + 8 * half;
        int col = n0 + wn * 32 + j * 16 + nn;
        float v = acc[i][j][r];
        if (BF16OUT) ((u16*)Cout)[(size_t)row * N + col] = f2bf(v);
        else         ((float*)Cout)[(size_t)row * N + col] = v;
      }
}

// ---------------------------------------------------------------- beta / g gates
__global__ __launch_bounds__(256) void gate_kernel(
    const float* __restrict__ x, const float* __restrict__ Wba,
    const float* __restrict__ dt_bias, const float* __restrict__ A_log,
    float* __restrict__ gbuf, float* __restrict__ bbuf)
{
  const int row = blockIdx.x;                 // token index in [0, B*T)
  const int t = threadIdx.x;
  __shared__ float xr[1024];
#pragma unroll
  for (int e = 0; e < 4; ++e) xr[t * 4 + e] = x[(size_t)row * 1024 + t * 4 + e];
  __syncthreads();
  if (t < 16) {
    float s = 0.f;
    for (int i = 0; i < 1024; ++i) s += xr[i] * Wba[i * 16 + t];
    if (t < 8) {
      bbuf[(size_t)row * 8 + t] = 1.f / (1.f + expf(-s));
    } else {
      int h = t - 8;
      float a = s + dt_bias[h];
      float sp = (a > 20.f) ? a : log1pf(expf(a));    // softplus
      gbuf[(size_t)row * 8 + h] = -expf(A_log[h]) * sp;
    }
  }
}

// ---------------------------------------------------------------- causal depthwise conv + SiLU
__global__ __launch_bounds__(256) void conv_silu_kernel(
    const u16* __restrict__ qkvz, const float* __restrict__ w,
    const float* __restrict__ bias, u16* __restrict__ qkvc)
{
  const int row = blockIdx.x;
  const int b = row >> 12, tt = row & 4095;   // T = 4096
#pragma unroll
  for (int cc = 0; cc < 12; ++cc) {
    int c = threadIdx.x + cc * 256;           // channel in [0,3072)
    float acc = bias[c];
#pragma unroll
    for (int j = 0; j < 4; ++j) {
      int ts = tt - 3 + j;
      if (ts >= 0)
        acc += bf2f(qkvz[((size_t)(b * 4096 + ts)) * 4096 + c]) * w[c * 4 + j];
    }
    float y = acc / (1.f + expf(-acc));       // silu
    qkvc[(size_t)row * 3072 + c] = f2bf(y);
  }
}

// ---------------------------------------------------------------- chunked gated delta rule
// One block per (b,h). Chunk length 16. State S[128x128] f32 lives in 8 v8f accumulators
// per wave (wave w owns dk rows [16w,16w+16)); mirrored to LDS as bf16 S^T for WMMA reuse.
// Chunk tiles staged via TDM (tensor_load_to_lds) when available.
__global__ __launch_bounds__(256) void delta_rule_kernel(
    const u16* __restrict__ qkvc,   // [B,T,3072] bf16
    const float* __restrict__ gbuf, // [B,T,8]
    const float* __restrict__ bbuf, // [B,T,8]
    float* __restrict__ core)       // [B,T,1024] f32
{
  constexpr int T = 4096, CH = 16, DK = 128, DV = 128;
  const float SCALE = 0.08838834764831845f;   // dk^-0.5

  __shared__ __align__(16) u16 s0T[DV][DK + 8];     // S^T (dv x dk) bf16
  __shared__ __align__(16) u16 qlds[CH][DK + 8];
  __shared__ __align__(16) u16 klds[CH][DK + 8];
  __shared__ __align__(16) u16 vbf[CH][DK + 8];     // v, bf16
  __shared__ float ks0[CH][DV + 4];                 // K @ S0
  __shared__ float qs0[CH][DV + 4];                 // Q @ S0
  __shared__ float KKs[CH][CH], QKs[CH][CH], Mm[CH][CH];
  __shared__ __align__(4) u16 attn[CH][40];         // masked decayed QK^T (K padded to 32)
  __shared__ __align__(4) u16 ubfT[DV][40];         // U^T (dv x chunk), padded
  __shared__ __align__(4) u16 kpT[DK][40];          // (k_i * exp(Gc15-Gc_i))^T, padded
  __shared__ float Graw[CH], Gc[CH], egl[CH], bl[CH];

  const int b = blockIdx.x >> 3, h = blockIdx.x & 7;
  const int t = threadIdx.x, lane = t & 31, wave = t >> 5, half = lane >> 4;

  v8f accS[8] = {};   // wave's 16 dk-rows x 128 dv of S

  for (int idx = t; idx < DV * (DK + 8); idx += 256) ((u16*)s0T)[idx] = 0;
  for (int idx = t; idx < DV * 40; idx += 256) { ((u16*)ubfT)[idx] = 0; ((u16*)kpT)[idx] = 0; }
  for (int idx = t; idx < CH * 40; idx += 256) ((u16*)attn)[idx] = 0;
  __syncthreads();

  for (int cidx = 0; cidx < T / CH; ++cidx) {
    const int t0 = cidx * CH;
    const size_t rowbase = (size_t)b * T + t0;

#if HAVE_TDM
    if (wave == 0) {                    // TDM: three 16x128 bf16 tiles, row stride 3072
      const u16* gq = qkvc + rowbase * 3072 + h * DK;
      tdm_load_tile(lds_off(&qlds[0][0]), gq);
      tdm_load_tile(lds_off(&klds[0][0]), gq + 1024);
      tdm_load_tile(lds_off(&vbf[0][0]),  gq + 2048);
      wait_tensor0();
    }
#else
    { // stage q,k,v chunk [16 x 128] each (bf16)
      int i = t >> 4, c0 = (t & 15) << 3;
      const u16* gq = qkvc + (rowbase + i) * 3072 + h * DK + c0;
      *(uint4*)&qlds[i][c0] = *(const uint4*)gq;
      *(uint4*)&klds[i][c0] = *(const uint4*)(gq + 1024);
      *(uint4*)&vbf[i][c0]  = *(const uint4*)(gq + 2048);
    }
#endif
    if (t < CH) {
      Graw[t] = gbuf[(rowbase + t) * 8 + h];
      bl[t]   = bbuf[(rowbase + t) * 8 + h];
    }
    __syncthreads();

    if (t == 0) {                       // inclusive cumsum of g, and exp()
      float s = 0.f;
      for (int i = 0; i < CH; ++i) { s += Graw[i]; Gc[i] = s; egl[i] = expf(s); }
    }

    if (wave < 2) {                     // KK^T (wave0) / QK^T (wave1): [16,128]@[128,16]
      const u16* ab = (wave == 0) ? &klds[0][0] : &qlds[0][0];
      v8f a16 = {};
#pragma unroll
      for (int s = 0; s < 4; ++s) {
        v16bf af  = ld_fragA(ab + s * 32, DK + 8, 0);
        v16bf bfv = ld_fragB(&klds[0][0] + s * 32, DK + 8, 0);
        a16 = wmma_bf16(af, bfv, a16);
      }
      float (*dst)[CH] = (wave == 0) ? KKs : QKs;
#pragma unroll
      for (int r = 0; r < 8; ++r) dst[r + 8 * half][lane & 15] = a16[r];
    }

    { // K@S0 and Q@S0: [16,128]@[128,128]; wave w computes dv tile w
      v8f aK = {}, aQ = {};
#pragma unroll
      for (int s = 0; s < 4; ++s) {
        v16bf bS  = ld_fragB(&s0T[0][0] + s * 32, DK + 8, wave * 16);
        v16bf afK = ld_fragA(&klds[0][0] + s * 32, DK + 8, 0);
        v16bf afQ = ld_fragA(&qlds[0][0] + s * 32, DK + 8, 0);
        aK = wmma_bf16(afK, bS, aK);
        aQ = wmma_bf16(afQ, bS, aQ);
      }
#pragma unroll
      for (int r = 0; r < 8; ++r) {
        int i = r + 8 * half, d = wave * 16 + (lane & 15);
        ks0[i][d] = aK[r];
        qs0[i][d] = aQ[r];
      }
    }
    __syncthreads();

    { // M (strictly lower) and decayed masked attention scores (bf16)
      int i = t >> 4, j = t & 15;
      float dec = expf(Gc[i] - Gc[j]);
      Mm[i][j]   = (j < i)  ? bl[i] * dec * KKs[i][j] : 0.f;
      attn[i][j] = f2bf((j <= i) ? SCALE * dec * QKs[i][j] : 0.f);
    }
    __syncthreads();

    if (t < DV) {                       // forward substitution: (I+M) U = B(V - e^G K S0)
      float u[CH];
#pragma unroll
      for (int i = 0; i < CH; ++i) {
        float r = bl[i] * (bf2f(vbf[i][t]) - egl[i] * ks0[i][t]);
#pragma unroll
        for (int j = 0; j < i; ++j) r -= Mm[i][j] * u[j];
        u[i] = r;
        ubfT[t][i] = f2bf(r);
      }
    } else {                            // Kp^T: k_i * exp(Gc15 - Gc_i), transposed
      int d = t - 128;
      float g15 = Gc[CH - 1];
#pragma unroll
      for (int i = 0; i < CH; ++i)
        kpT[d][i] = f2bf(bf2f(klds[i][d]) * expf(g15 - Gc[i]));
    }
    __syncthreads();

    { // O = scale*e^G * (Q@S0) + Attn @ U ; wave handles dv tile `wave`
      v8f ao = {};
      v16bf af  = ld_fragA(&attn[0][0], 40, 0);
      v16bf bfu = ld_fragB(&ubfT[0][0], 40, wave * 16);
      ao = wmma_bf16(af, bfu, ao);
#pragma unroll
      for (int r = 0; r < 8; ++r) {
        int i = r + 8 * half, d = wave * 16 + (lane & 15);
        core[(rowbase + i) * 1024 + h * DV + d] = SCALE * egl[i] * qs0[i][d] + ao[r];
      }
    }

    { // S <- e^G15 * S + Kp^T @ U  (rank-16 update via WMMA, K padded to 32)
      float dec = egl[CH - 1];
      v16bf aS = ld_fragA(&kpT[0][0], 40, wave * 16);
#pragma unroll
      for (int nt = 0; nt < 8; ++nt) {
        v16bf bU = ld_fragB(&ubfT[0][0], 40, nt * 16);
        accS[nt] = accS[nt] * dec;
        accS[nt] = wmma_bf16(aS, bU, accS[nt]);
      }
#pragma unroll
      for (int nt = 0; nt < 8; ++nt)    // publish S^T bf16 for next chunk
#pragma unroll
        for (int r = 0; r < 8; ++r) {
          int m = wave * 16 + r + 8 * half;
          int n = nt * 16 + (lane & 15);
          s0T[n][m] = f2bf(accS[nt][r]);
        }
    }
    __syncthreads();
  }
}

// ---------------------------------------------------------------- gated RMSNorm
__global__ __launch_bounds__(256) void rmsgate_kernel(
    const float* __restrict__ core, const u16* __restrict__ qkvz,
    const float* __restrict__ norm_w, u16* __restrict__ fused)
{
  const int row = blockIdx.x;
  const int t = threadIdx.x;
  __shared__ float sums[256];
  float cv[4]; float ss = 0.f;
#pragma unroll
  for (int e = 0; e < 4; ++e) {
    cv[e] = core[(size_t)row * 1024 + t * 4 + e];
    ss += cv[e] * cv[e];
  }
  sums[t] = ss;
  __syncthreads();
  for (int s = 128; s > 0; s >>= 1) {
    if (t < s) sums[t] += sums[t + s];
    __syncthreads();
  }
  float rms = rsqrtf(sums[0] / 1024.f + 1e-6f);
#pragma unroll
  for (int e = 0; e < 4; ++e) {
    int d = t * 4 + e;
    float z = bf2f(qkvz[(size_t)row * 4096 + 3072 + d]);
    float sz = z / (1.f + expf(-z));
    fused[(size_t)row * 1024 + d] = f2bf(cv[e] * rms * norm_w[d] * sz);
  }
}

// ---------------------------------------------------------------- launcher
extern "C" void kernel_launch(void* const* d_in, const int* in_sizes, int n_in,
                              void* d_out, int out_size, void* d_ws, size_t ws_size,
                              hipStream_t stream) {
  (void)in_sizes; (void)n_in; (void)out_size; (void)ws_size;
  const float* x       = (const float*)d_in[0];
  const float* W_qkvz  = (const float*)d_in[1];
  const float* W_ba    = (const float*)d_in[2];
  const float* conv_w  = (const float*)d_in[3];
  const float* conv_b  = (const float*)d_in[4];
  const float* dt_bias = (const float*)d_in[5];
  const float* A_log   = (const float*)d_in[6];
  const float* norm_w  = (const float*)d_in[7];
  const float* W_out   = (const float*)d_in[8];

  const int B = 4, T = 4096, D = 1024, BT = B * T;

  char* ws = (char*)d_ws;
  size_t off = 0;
  auto alloc = [&](size_t bytes) { void* p = ws + off; off += (bytes + 255) & ~(size_t)255; return p; };

  u16*   xb    = (u16*)  alloc((size_t)BT * D * 2);      //  32 MB
  u16*   wqb   = (u16*)  alloc((size_t)D * 4096 * 2);    //   8 MB
  u16*   wob   = (u16*)  alloc((size_t)D * 1024 * 2);    //   2 MB
  u16*   qkvz  = (u16*)  alloc((size_t)BT * 4096 * 2);   // 128 MB
  u16*   qkvc  = (u16*)  alloc((size_t)BT * 3072 * 2);   //  96 MB
  float* gb    = (float*)alloc((size_t)BT * 8 * 4);
  float* bb    = (float*)alloc((size_t)BT * 8 * 4);
  float* core  = (float*)alloc((size_t)BT * 1024 * 4);   //  64 MB
  u16*   fused = (u16*)  alloc((size_t)BT * 1024 * 2);   //  32 MB

  auto cvt = [&](const float* in, u16* out, size_t n) {
    f2bf_kernel<<<(unsigned)((n + 255) / 256), 256, 0, stream>>>(in, out, n);
  };
  cvt(x, xb, (size_t)BT * D);
  cvt(W_qkvz, wqb, (size_t)D * 4096);
  cvt(W_out, wob, (size_t)D * 1024);

  gemm_wmma_kernel<true ><<<dim3(4096 / 128, BT / 128), 256, 0, stream>>>(xb, wqb, qkvz, BT, 4096, D);
  gate_kernel     <<<BT, 256, 0, stream>>>(x, W_ba, dt_bias, A_log, gb, bb);
  conv_silu_kernel<<<BT, 256, 0, stream>>>(qkvz, conv_w, conv_b, qkvc);
  delta_rule_kernel<<<32, 256, 0, stream>>>(qkvc, gb, bb, core);
  rmsgate_kernel  <<<BT, 256, 0, stream>>>(core, qkvz, norm_w, fused);
  gemm_wmma_kernel<false><<<dim3(1024 / 128, BT / 128), 256, 0, stream>>>(fused, wob, d_out, BT, 1024, D);
}
